// ContrastiveLearner_77919296684692
// MI455X (gfx1250) — matrix operand order repaired and checked by reading, not compiled, not run on hardware
//
#include <hip/hip_runtime.h>
#include <cstdint>

typedef _Float16 v16h __attribute__((ext_vector_type(16)));
typedef _Float16 v8h  __attribute__((ext_vector_type(8)));
typedef float    v8f  __attribute__((ext_vector_type(8)));

#define LDSTRIDE 136   // 128 halves + 8 halves pad (16B) -> breaks 256B bank period

// ---- Tensor Data Mover: builtin exists on both toolchains, arity differs ----
//   ROCm 7.2 / clang-22 : 5 args (u32x4, i32x8, i32x4, i32x4, i32)
//   amdgpu-toolchain / clang-23+: 6 args (u32x4, i32x8, i32x4, i32x4, i32x8, i32)
#if defined(__HIP_DEVICE_COMPILE__)
#if __has_builtin(__builtin_amdgcn_tensor_load_to_lds)
#define USE_TDM 1
typedef unsigned int u32x4 __attribute__((ext_vector_type(4)));
typedef int          i32x4 __attribute__((ext_vector_type(4)));
typedef int          i32x8 __attribute__((ext_vector_type(8)));
#endif
#endif

// Native f32 global atomic add (avoid CAS-loop lowering of atomicAdd(float*))
__device__ __forceinline__ void atomAddF(float* p, float v) {
#if defined(__HIP_DEVICE_COMPILE__)
    unsafeAtomicAdd(p, v);
#else
    (void)p; (void)v;
#endif
}

// ---------------------------------------------------------------------------
// 0) zero the accumulators used across kernels
__global__ void init_kernel(float* __restrict__ rowSum, float* __restrict__ lossAcc, int total) {
    int i = blockIdx.x * 256 + threadIdx.x;
    if (i < total) rowSum[i] = 0.0f;
    if (i < 2)     lossAcc[i] = 0.0f;
}

// ---------------------------------------------------------------------------
// 1) stable prefix-scan over nlabel to build the gather permutation
//    perm[0..n)  = rows with nlabel[:,0]==1 (users)
//    perm[n..2n) = rows with nlabel[:,1]==1 (items)
__global__ void scan_kernel(const int* __restrict__ nlabel, int* __restrict__ perm, int N) {
    const int T = 1024;
    int t = threadIdx.x;
    int per = N / T;                 // 16
    __shared__ int s0[1024], s1[1024];
    int c0 = 0, c1 = 0;
    int base = t * per;
    for (int e = 0; e < per; ++e) {
        c0 += (nlabel[2 * (base + e) + 0] != 0);
        c1 += (nlabel[2 * (base + e) + 1] != 0);
    }
    s0[t] = c0; s1[t] = c1;
    __syncthreads();
    for (int off = 1; off < T; off <<= 1) {     // Hillis–Steele inclusive scan
        int a0 = (t >= off) ? s0[t - off] : 0;
        int a1 = (t >= off) ? s1[t - off] : 0;
        __syncthreads();
        s0[t] += a0; s1[t] += a1;
        __syncthreads();
    }
    int ex0 = s0[t] - c0, ex1 = s1[t] - c1;     // exclusive offsets
    int n = N / 2;
    for (int e = 0; e < per; ++e) {
        int idx = base + e;
        if (nlabel[2 * idx + 0] != 0) perm[ex0++] = idx;
        if (nlabel[2 * idx + 1] != 0) perm[n + ex1++] = idx;
    }
}

// ---------------------------------------------------------------------------
// 2) row-normalize emb1 -> z1h (f16) and emb2[perm] -> z2h (f16)
//    one wave per row: 32 lanes x float4 = 128 floats
__global__ void normalize_kernel(const float* __restrict__ emb1,
                                 const float* __restrict__ emb2,
                                 const int*  __restrict__ perm,
                                 _Float16* __restrict__ z1h,
                                 _Float16* __restrict__ z2h, int N) {
    int wave = threadIdx.x >> 5, lane = threadIdx.x & 31;
    int r = blockIdx.x * 8 + wave;
    if (r >= N) return;

    // z1 from emb1[r]
    {
        const float4* src = (const float4*)(emb1 + (size_t)r * 128);
        float4 v = src[lane];
        float ss = v.x * v.x + v.y * v.y + v.z * v.z + v.w * v.w;
        for (int m = 1; m < 32; m <<= 1) ss += __shfl_xor(ss, m, 32);
        float sc = 1.0f / fmaxf(sqrtf(ss), 1e-12f);
        _Float16* dst = z1h + (size_t)r * 128 + lane * 4;
        dst[0] = (_Float16)(v.x * sc); dst[1] = (_Float16)(v.y * sc);
        dst[2] = (_Float16)(v.z * sc); dst[3] = (_Float16)(v.w * sc);
    }
    // z2 from emb2[perm[r]]
    {
        int pr = perm[r];
        const float4* src = (const float4*)(emb2 + (size_t)pr * 128);
        float4 v = src[lane];
        float ss = v.x * v.x + v.y * v.y + v.z * v.z + v.w * v.w;
        for (int m = 1; m < 32; m <<= 1) ss += __shfl_xor(ss, m, 32);
        float sc = 1.0f / fmaxf(sqrtf(ss), 1e-12f);
        _Float16* dst = z2h + (size_t)r * 128 + lane * 4;
        dst[0] = (_Float16)(v.x * sc); dst[1] = (_Float16)(v.y * sc);
        dst[2] = (_Float16)(v.z * sc); dst[3] = (_Float16)(v.w * sc);
    }
}

// ---------------------------------------------------------------------------
// 3) fused InfoNCE GEMM: sim = (z2 @ z1^T)/tau, never materialized.
//    grid (n/128, n/128, 2); block 256 = 8 waves.
//    wave w: rows m0+16w..+15, all 8 column tiles (A frag reused 8x per k-step)
//    B panel staged into LDS by the Tensor Data Mover with hardware row padding.
__global__ __launch_bounds__(256) void infonce_kernel(
        const _Float16* __restrict__ z1h, const _Float16* __restrict__ z2h,
        const float* __restrict__ tau,
        float* __restrict__ rowSum, float* __restrict__ posBuf, int n) {
    __shared__ __align__(16) _Float16 Bs[128 * LDSTRIDE];   // B panel, padded rows

    const int z  = blockIdx.z;                  // 0 = user group, 1 = item group
    const int m0 = blockIdx.x * 128;
    const int n0 = blockIdx.y * 128;
    const _Float16* Arows = z2h + (size_t)z * n * 128;   // A = z2
    const _Float16* Brows = z1h + (size_t)z * n * 128;   // B columns = z1 rows

    const int tid = threadIdx.x, lane = tid & 31, wave = tid >> 5;
    const int hi = lane >> 4, l15 = lane & 15;

    // ---- stage B panel (128 rows x 128 halves) into LDS with +16B/row pad ----
#ifdef USE_TDM
    if (wave == 0) {
        unsigned long long ga =
            (unsigned long long)(uintptr_t)(Brows + (size_t)n0 * 128);
        // D# group0: count=1, lds_addr=0 (Bs is the only LDS alloc), global addr, type=2
        u32x4 g0 = { 1u, 0u,
                     (unsigned)(ga & 0xFFFFFFFFu),
                     (unsigned)((ga >> 32) & 0x01FFFFFFu) | 0x80000000u };
        // D# group1: data_size=2B, pad_enable, pad_interval=64DW(row), pad_amount=4DW(16B)
        i32x8 g1 = { (1 << 16) | (1 << 20) | (5 << 22) | (3 << 25),
                     (128 << 16),            // tensor_dim0 = 128
                     (128 << 16),            // tensor_dim1 = 128 (tile-sized; no OOB)
                     (128 << 16),            // tile_dim0 = 128
                     128,                    // tile_dim1 = 128, tile_dim2 = 0
                     128,                    // tensor_dim0_stride = 128
                     0, 0 };
        i32x4 gz = { 0, 0, 0, 0 };           // 2D tensor: groups 2/3 unused
#if __clang_major__ >= 23
        i32x8 gz8 = { 0, 0, 0, 0, 0, 0, 0, 0 };
        __builtin_amdgcn_tensor_load_to_lds(g0, g1, gz, gz, gz8, 0);
#else
        __builtin_amdgcn_tensor_load_to_lds(g0, g1, gz, gz, 0);
#endif
        __builtin_amdgcn_s_wait_tensorcnt(0);
        // Plant a REAL store to Bs so GlobalOpt cannot prove "never stored ->
        // loads fold to undef". Volatile RMW of element 0 with its own value:
        // benign at runtime, decisive in IR.
        if (lane == 0) {
            volatile _Float16* vb = Bs;
            vb[0] = vb[0];
        }
    }
    // Escape Bs's address into an asm that may write memory: with the pointer
    // captured, alias analysis must assume the staging point writes the panel.
    // (A bare memory clobber was NOT enough: Bs's address never escaped, so
    // GlobalOpt still folded all its loads to undef -> wmma collapsed 32 -> 4.)
    {
        const _Float16* esc = Bs;
        asm volatile("" :: "v"(esc) : "memory");
    }
#else
    for (int i = tid; i < 128 * 16; i += 256) {          // 16 uint4 per row
        int row = i >> 4, c = i & 15;
        *(uint4*)(&Bs[row * LDSTRIDE + c * 8]) =
            *(const uint4*)(Brows + (size_t)(n0 + row) * 128 + c * 8);
    }
#endif
    __syncthreads();

    const float inv_tau = 1.0f / tau[0];
    const int mrow = m0 + wave * 16;

    v8f acc[8];
    const v8f vzero = { 0.f, 0.f, 0.f, 0.f, 0.f, 0.f, 0.f, 0.f };
    for (int t = 0; t < 8; ++t) acc[t] = vzero;

    for (int k0 = 0; k0 < 128; k0 += 32) {
        // A fragment (16x32 f16): lane hi=0 -> K {0..7},{16..23}; hi=1 -> {8..15},{24..31}
        const _Float16* ap = Arows + (size_t)(mrow + l15) * 128 + k0 + hi * 8;
        v8h alo = *(const v8h*)(ap);
        v8h ahi = *(const v8h*)(ap + 16);
        v16h a = __builtin_shufflevector(alo, ahi, 0, 1, 2, 3, 4, 5, 6, 7,
                                         8, 9, 10, 11, 12, 13, 14, 15);
        for (int t = 0; t < 8; ++t) {
            // B fragment (32x16): lane holds column (16t+l15), hi=0 -> K {0..7},{16..23}; hi=1 -> {8..15},{24..31}
            const _Float16* bp = &Bs[(16 * t + l15) * LDSTRIDE + k0 + hi * 8];
            v8h blo = *(const v8h*)(bp);
            v8h bhi = *(const v8h*)(bp + 16);
            v16h b = __builtin_shufflevector(blo, bhi, 0, 1, 2, 3, 4, 5, 6, 7,
                                             8, 9, 10, 11, 12, 13, 14, 15);
            acc[t] = __builtin_amdgcn_wmma_f32_16x16x32_f16(
                         false, a, false, b, (short)0, acc[t], false, false);
        }
    }

    // ---- fused epilogue: scale, exp, diagonal capture, row sums ----
    for (int v = 0; v < 8; ++v) {
        int row = mrow + v + hi * 8;                  // row within group
        float rs = 0.0f;
        for (int t = 0; t < 8; ++t) {
            int col = n0 + 16 * t + l15;
            float s = acc[t][v] * inv_tau;
            rs += __expf(s);
            if (col == row) posBuf[(size_t)z * n + row] = s;   // unique writer
        }
        // reduce the 16 lanes sharing this row (half-wave xor tree)
        rs += __shfl_xor(rs, 1, 32);
        rs += __shfl_xor(rs, 2, 32);
        rs += __shfl_xor(rs, 4, 32);
        rs += __shfl_xor(rs, 8, 32);
        if (l15 == 0)
            atomAddF(&rowSum[(size_t)z * n + row], rs);
    }
}

// ---------------------------------------------------------------------------
// 4) per-row loss terms -> two group accumulators
__global__ void loss_kernel(const float* __restrict__ rowSum,
                            const float* __restrict__ posBuf,
                            float* __restrict__ lossAcc, int n) {
    int r = blockIdx.x * 256 + threadIdx.x;            // over 2n rows
    __shared__ float red[256];
    float p = posBuf[r];
    float denom = rowSum[r] - expf(p);                 // sum over off-diagonal
    float val = logf(denom + 1e-8f) - p;
    red[threadIdx.x] = val;
    __syncthreads();
    for (int s = 128; s > 0; s >>= 1) {
        if (threadIdx.x < s) red[threadIdx.x] += red[threadIdx.x + s];
        __syncthreads();
    }
    if (threadIdx.x == 0)
        atomAddF(&lossAcc[(r < n) ? 0 : 1], red[0]);   // n % 256 == 0 -> uniform
}

__global__ void finalize_kernel(const float* __restrict__ lossAcc,
                                float* __restrict__ out, int n) {
    out[0] = 0.5f * (lossAcc[0] + lossAcc[1]) / (float)n;
}

// ---------------------------------------------------------------------------
// 5) MLP head: edges[r] = [emb1[r], emb2[perm[r]], emb1[r+n], emb2[perm[r+n]]]
//    h = leaky_relu(edges@W1+b1, 0.2); preds = sigmoid(h@W2+b2)
__global__ void fc_kernel(const float* __restrict__ emb1, const float* __restrict__ emb2,
                          const int* __restrict__ perm,
                          const float* __restrict__ W1, const float* __restrict__ b1,
                          const float* __restrict__ W2, const float* __restrict__ b2,
                          float* __restrict__ preds, int n) {
    int r = blockIdx.x, t = threadIdx.x;               // 64 threads
    __shared__ float eds[512];
    __shared__ float h[64];
    const float* s0 = emb1 + (size_t)r * 128;
    const float* s1 = emb2 + (size_t)perm[r] * 128;
    const float* s2 = emb1 + (size_t)(r + n) * 128;
    const float* s3 = emb2 + (size_t)perm[r + n] * 128;
    for (int i = t; i < 128; i += 64) {
        eds[i] = s0[i]; eds[128 + i] = s1[i];
        eds[256 + i] = s2[i]; eds[384 + i] = s3[i];
    }
    __syncthreads();
    float acc = b1[t];
    for (int k = 0; k < 512; ++k) acc = fmaf(eds[k], W1[k * 64 + t], acc);
    h[t] = (acc > 0.0f) ? acc : 0.2f * acc;
    __syncthreads();
    if (t == 0) {
        float a2 = b2[0];
        for (int j = 0; j < 64; ++j) a2 = fmaf(h[j], W2[j], a2);
        preds[r] = 1.0f / (1.0f + expf(-a2));
    }
}

// ---------------------------------------------------------------------------
extern "C" void kernel_launch(void* const* d_in, const int* in_sizes, int n_in,
                              void* d_out, int out_size, void* d_ws, size_t ws_size,
                              hipStream_t stream) {
    const float* emb1   = (const float*)d_in[0];
    const float* emb2   = (const float*)d_in[1];
    const int*   nlabel = (const int*)d_in[2];
    const float* tau    = (const float*)d_in[3];
    const float* W1     = (const float*)d_in[4];
    const float* b1     = (const float*)d_in[5];
    const float* W2     = (const float*)d_in[6];
    const float* b2     = (const float*)d_in[7];

    const int N = in_sizes[0] / 128;   // 16384
    const int n = N / 2;               // 8192

    char* ws = (char*)d_ws;
    int*      perm    = (int*)ws;                               // 64 KB
    float*    rowSum  = (float*)(ws + 65536);                   // 64 KB
    float*    posBuf  = (float*)(ws + 131072);                  // 64 KB
    float*    lossAcc = (float*)(ws + 196608);                  // 8 B
    _Float16* z1h     = (_Float16*)(ws + 262144);               // 4 MB
    _Float16* z2h     = (_Float16*)(ws + 262144 + (size_t)N * 128 * 2);
    float*    out     = (float*)d_out;

    init_kernel<<<(N + 255) / 256, 256, 0, stream>>>(rowSum, lossAcc, N);
    scan_kernel<<<1, 1024, 0, stream>>>(nlabel, perm, N);
    normalize_kernel<<<N / 8, 256, 0, stream>>>(emb1, emb2, perm, z1h, z2h, N);

    dim3 grid(n / 128, n / 128, 2);
    infonce_kernel<<<grid, 256, 0, stream>>>(z1h, z2h, tau, rowSum, posBuf, n);

    loss_kernel<<<N / 256, 256, 0, stream>>>(rowSum, posBuf, lossAcc, n);
    finalize_kernel<<<1, 1, 0, stream>>>(lossAcc, out, n);
    fc_kernel<<<n, 64, 0, stream>>>(emb1, emb2, perm, W1, b1, W2, b2, out + 1, n);
}
